// SpaceSelfAttentionModality_3152505995921
// MI455X (gfx1250) — matrix operand order repaired and checked by reading, not compiled, not run on hardware
//
#include <hip/hip_runtime.h>

// ---------------------------------------------------------------------------
// Types / helpers
// ---------------------------------------------------------------------------
typedef __attribute__((ext_vector_type(16))) __bf16 bf16x16;
typedef __attribute__((ext_vector_type(8)))  float  f32x8;

union FragU {
  bf16x16 v;
  uint4   q[2];
  unsigned short us[16];
};

__device__ __forceinline__ unsigned short f2bf(float f) {
  unsigned int u = __builtin_bit_cast(unsigned int, f);
  unsigned int r = u + 0x7FFFu + ((u >> 16) & 1u);   // round-to-nearest-even
  return (unsigned short)(r >> 16);
}

__device__ __forceinline__ f32x8 wmma_bf16(const FragU& a, const FragU& b, f32x8 c) {
  // D = A(16x32 bf16) * B(32x16 bf16) + C(16x16 f32)
  return __builtin_amdgcn_wmma_f32_16x16x32_bf16(false, a.v, false, b.v,
                                                 (short)0, c, false, false);
}

// Problem constants
#define NBT   8          // B*T
#define SEQL  1024
#define DMOD  1024
#define NH    16
#define HDIM  64

// Workspace layout (bytes). Total ~92 MB (fits comfortably in 192 MB L2).
static const size_t OFF_XB   = 0;                                      // x   bf16 [8192][1024]
static const size_t OFF_WQKV = OFF_XB   + (size_t)8192*1024*2;         // WqkvT bf16 [3072][1024] (n-major, k-inner)
static const size_t OFF_BIAS = OFF_WQKV + (size_t)3072*1024*2;         // bias f32 [3072]
static const size_t OFF_WOT  = OFF_BIAS + (size_t)3072*4;              // WoT bf16 [1024][1024] (o-major, j-inner)
static const size_t OFF_Q    = OFF_WOT  + (size_t)1024*1024*2;         // Q  bf16 [128][1024][64]
static const size_t OFF_K    = OFF_Q    + (size_t)128*1024*64*2;       // K  bf16 [128][1024][64]
static const size_t OFF_VT   = OFF_K    + (size_t)128*1024*64*2;       // Vt bf16 [128][64][1024]
static const size_t OFF_OB   = OFF_VT   + (size_t)128*64*1024*2;       // ctx bf16 [8192][1024]

// ---------------------------------------------------------------------------
// Prep kernels: f32 -> bf16 conversions and operand transposes
// ---------------------------------------------------------------------------
__global__ void cvt_x_kernel(const float* __restrict__ x, unsigned short* __restrict__ xb, int n) {
  int i = blockIdx.x * blockDim.x + threadIdx.x;
  if (i < n) xb[i] = f2bf(x[i]);
}

// WqkvT[n][d] : n in [0,3072). Wq/Wk/Wv flat are [d][h*64+k] = [d][n%1024].
// Fold 1/sqrt(HDIM) = 0.125 into the Q slice.
__global__ void build_wqkvT_kernel(const float* __restrict__ Wq, const float* __restrict__ Wk,
                                   const float* __restrict__ Wv, unsigned short* __restrict__ WT) {
  int idx = blockIdx.x * blockDim.x + threadIdx.x;       // idx = n*1024 + d
  if (idx >= 3072 * 1024) return;
  int n = idx >> 10;
  int d = idx & 1023;
  float v;
  if (n < 1024)       v = 0.125f * Wq[d * 1024 + n];
  else if (n < 2048)  v = Wk[d * 1024 + (n - 1024)];
  else                v = Wv[d * 1024 + (n - 2048)];
  WT[idx] = f2bf(v);
}

__global__ void build_bias_kernel(const float* __restrict__ bq, const float* __restrict__ bk,
                                  const float* __restrict__ bv, float* __restrict__ bias) {
  int i = blockIdx.x * blockDim.x + threadIdx.x;
  if (i >= 3072) return;
  if (i < 1024)      bias[i] = 0.125f * bq[i];
  else if (i < 2048) bias[i] = bk[i - 1024];
  else               bias[i] = bv[i - 2048];
}

// WoT[o][j] = Wo[h][d][o] with j = h*64+d  (Wo flat = j*1024 + o)
__global__ void build_woT_kernel(const float* __restrict__ Wo, unsigned short* __restrict__ WoT) {
  int idx = blockIdx.x * blockDim.x + threadIdx.x;       // idx = o*1024 + j
  if (idx >= 1024 * 1024) return;
  int o = idx >> 10;
  int j = idx & 1023;
  WoT[idx] = f2bf(Wo[j * 1024 + o]);
}

// ---------------------------------------------------------------------------
// Kernel 1: QKV projection.  [8192 x 1024] x [1024 x 3072] -> Q,K (s,d) + Vt (d,s)
// 256 threads = 8 waves (2 M x 4 N), wave tile 64x64, WG tile 128x256.
// ---------------------------------------------------------------------------
__global__ __launch_bounds__(256) void qkv_gemm_kernel(
    const unsigned short* __restrict__ xb, const unsigned short* __restrict__ WT,
    const float* __restrict__ bias,
    unsigned short* __restrict__ Qb, unsigned short* __restrict__ Kb,
    unsigned short* __restrict__ Vt) {
  int tid = threadIdx.x;
  int lane = tid & 31, w = tid >> 5;
  int ln = lane & 15, hi = lane >> 4;
  int wm = w >> 2, wn = w & 3;
  int m0 = blockIdx.x * 128 + wm * 64;
  int n0 = blockIdx.y * 256 + wn * 64;

  f32x8 acc[4][4];
  for (int i = 0; i < 4; ++i)
    for (int j = 0; j < 4; ++j)
      for (int r = 0; r < 8; ++r) acc[i][j][r] = 0.0f;

  int abase = hi * 8;     // A fragment per-lane K base
  int bbase = hi * 8;     // B fragment per-lane K base

  for (int k = 0; k < 1024; k += 32) {
    FragU af[4], bf[4];
    for (int i = 0; i < 4; ++i) {
      const unsigned short* p = xb + (size_t)(m0 + 16 * i + ln) * 1024 + k + abase;
      af[i].q[0] = *(const uint4*)p;
      af[i].q[1] = *(const uint4*)(p + 16);
    }
    for (int j = 0; j < 4; ++j) {
      const unsigned short* p = WT + (size_t)(n0 + 16 * j + ln) * 1024 + k + bbase;
      bf[j].q[0] = *(const uint4*)p;
      bf[j].q[1] = *(const uint4*)(p + 16);
    }
    for (int i = 0; i < 4; ++i)
      for (int j = 0; j < 4; ++j)
        acc[i][j] = wmma_bf16(af[i], bf[j], acc[i][j]);
  }

  // Epilogue: add bias, convert to bf16, scatter to Q / K / Vt layouts.
  for (int j = 0; j < 4; ++j) {
    int n = n0 + 16 * j + ln;
    float bv_ = bias[n];
    for (int i = 0; i < 4; ++i) {
      for (int r = 0; r < 8; ++r) {
        int row = m0 + 16 * i + r + hi * 8;
        int b = row >> 10;
        int s = row & 1023;
        unsigned short hval = f2bf(acc[i][j][r] + bv_);
        if (n < 1024) {
          int hh = n >> 6, dd = n & 63;
          Qb[((size_t)(b * 16 + hh) << 16) + s * 64 + dd] = hval;
        } else if (n < 2048) {
          int n2 = n - 1024, hh = n2 >> 6, dd = n2 & 63;
          Kb[((size_t)(b * 16 + hh) << 16) + s * 64 + dd] = hval;
        } else {
          int n2 = n - 2048, hh = n2 >> 6, dd = n2 & 63;
          Vt[((size_t)(b * 16 + hh) << 16) + dd * 1024 + s] = hval;
        }
      }
    }
  }
}

// ---------------------------------------------------------------------------
// Kernel 2: flash attention with modality mask. One WG per (b,h), 8 waves,
// each wave owns 128 queries (8 tiles of 16). Online softmax; probs are
// transposed C->A fragment through a 1KB/wave LDS staging tile.
// ---------------------------------------------------------------------------
__global__ __launch_bounds__(256) void attn_kernel(
    const unsigned short* __restrict__ Qb, const unsigned short* __restrict__ Kb,
    const unsigned short* __restrict__ Vt, const int* __restrict__ modids,
    unsigned short* __restrict__ ob) {
  __shared__ int smod[1024];
  __shared__ unsigned short stage[8][16 * 32];   // 8 KB: per-wave prob transpose tile

  int tid = threadIdx.x;
  for (int i = tid; i < 1024; i += 256) smod[i] = modids[i];
  __syncthreads();

  int bh = blockIdx.x;
  int bb = bh >> 4, h = bh & 15;
  const unsigned short* Qp = Qb + ((size_t)bh << 16);
  const unsigned short* Kp = Kb + ((size_t)bh << 16);
  const unsigned short* Vp = Vt + ((size_t)bh << 16);

  int lane = tid & 31, w = tid >> 5;
  int ln = lane & 15, hi = lane >> 4;
  unsigned short* st = &stage[w][0];

  for (int qt = 0; qt < 8; ++qt) {
    int q0 = w * 128 + qt * 16;

    // Q A-fragments for d-chunks [0,32) and [32,64)
    FragU aq[2];
    for (int c = 0; c < 2; ++c) {
      const unsigned short* p = Qp + (size_t)(q0 + ln) * 64 + c * 32 + hi * 8;
      aq[c].q[0] = *(const uint4*)p;
      aq[c].q[1] = *(const uint4*)(p + 16);
    }

    // Per-row query modality (rows r + 8*hi of this tile)
    int   qm[8];
    int   qlat[8];
    for (int r = 0; r < 8; ++r) {
      int qrow = q0 + r + hi * 8;
      qm[r]   = smod[qrow];
      qlat[r] = (qrow < 64) ? 1 : 0;
    }

    float mrow[8], lrow[8];
    f32x8 accO[4];
    for (int r = 0; r < 8; ++r) { mrow[r] = -1e30f; lrow[r] = 0.0f; }
    for (int t = 0; t < 4; ++t)
      for (int r = 0; r < 8; ++r) accO[t][r] = 0.0f;

    for (int kb = 0; kb < 1024; kb += 32) {
      // ---- scores: two 16x16 tiles over 32 keys, contraction d=64 ----
      f32x8 s0, s1;
      for (int r = 0; r < 8; ++r) { s0[r] = 0.0f; s1[r] = 0.0f; }
      for (int t = 0; t < 2; ++t) {
        const unsigned short* krow = Kp + (size_t)(kb + t * 16 + ln) * 64;
        for (int c = 0; c < 2; ++c) {
          FragU kf;
          const unsigned short* p = krow + c * 32 + hi * 8;
          kf.q[0] = *(const uint4*)p;
          kf.q[1] = *(const uint4*)(p + 16);
          if (t == 0) s0 = wmma_bf16(aq[c], kf, s0);
          else        s1 = wmma_bf16(aq[c], kf, s1);
        }
      }

      int km0 = smod[kb + ln];
      int km1 = smod[kb + 16 + ln];

      // ---- online softmax update (per row r across the 16-lane half) ----
      for (int r = 0; r < 8; ++r) {
        bool a0 = qlat[r] || (qm[r] == km0);
        bool a1 = qlat[r] || (qm[r] == km1);
        float c0 = a0 ? s0[r] : -1e30f;
        float c1 = a1 ? s1[r] : -1e30f;
        float bm = fmaxf(c0, c1);
        for (int off = 1; off < 16; off <<= 1)
          bm = fmaxf(bm, __shfl_xor(bm, off, 32));
        float nm    = fmaxf(mrow[r], bm);
        float scale = __expf(mrow[r] - nm);
        mrow[r] = nm;
        float e0 = a0 ? __expf(s0[r] - nm) : 0.0f;
        float e1 = a1 ? __expf(s1[r] - nm) : 0.0f;
        float ps = e0 + e1;
        for (int off = 1; off < 16; off <<= 1)
          ps += __shfl_xor(ps, off, 32);
        lrow[r] = lrow[r] * scale + ps;
        for (int t = 0; t < 4; ++t) accO[t][r] *= scale;
        // stage unnormalized probs [16 rows][32 keys] row-major bf16
        st[(r + hi * 8) * 32 + ln]      = f2bf(e0);
        st[(r + hi * 8) * 32 + 16 + ln] = f2bf(e1);
      }

      // ---- P A-fragment from LDS (transpose: keys now within lane) ----
      FragU pf;
      const unsigned short* sp = st + ln * 32 + hi * 8;
      pf.q[0] = *(const uint4*)sp;
      pf.q[1] = *(const uint4*)(sp + 16);

      // ---- accumulate O += P * V  (4 d-tiles of 16) ----
      for (int t = 0; t < 4; ++t) {
        const unsigned short* vp = Vp + (size_t)(t * 16 + ln) * 1024 + kb + hi * 8;
        FragU vf;
        vf.q[0] = *(const uint4*)vp;
        vf.q[1] = *(const uint4*)(vp + 16);
        accO[t] = wmma_bf16(pf, vf, accO[t]);
      }
    }

    // ---- epilogue: normalize, store context bf16 [8192][1024] ----
    for (int t = 0; t < 4; ++t) {
      int dcol = t * 16 + ln;
      for (int r = 0; r < 8; ++r) {
        int qrow = q0 + r + hi * 8;
        float o = accO[t][r] * (1.0f / lrow[r]);
        ob[(size_t)(bb * 1024 + qrow) * 1024 + h * 64 + dcol] = f2bf(o);
      }
    }
  }
}

// ---------------------------------------------------------------------------
// Kernel 3: output projection  [8192 x 1024] x [1024 x 1024] + bo -> f32 out
// ---------------------------------------------------------------------------
__global__ __launch_bounds__(256) void out_gemm_kernel(
    const unsigned short* __restrict__ ob, const unsigned short* __restrict__ WoT,
    const float* __restrict__ bo, float* __restrict__ y) {
  int tid = threadIdx.x;
  int lane = tid & 31, w = tid >> 5;
  int ln = lane & 15, hi = lane >> 4;
  int wm = w >> 2, wn = w & 3;
  int m0 = blockIdx.x * 128 + wm * 64;
  int n0 = blockIdx.y * 256 + wn * 64;

  f32x8 acc[4][4];
  for (int i = 0; i < 4; ++i)
    for (int j = 0; j < 4; ++j)
      for (int r = 0; r < 8; ++r) acc[i][j][r] = 0.0f;

  for (int k = 0; k < 1024; k += 32) {
    FragU af[4], bf[4];
    for (int i = 0; i < 4; ++i) {
      const unsigned short* p = ob + (size_t)(m0 + 16 * i + ln) * 1024 + k + hi * 8;
      af[i].q[0] = *(const uint4*)p;
      af[i].q[1] = *(const uint4*)(p + 16);
    }
    for (int j = 0; j < 4; ++j) {
      const unsigned short* p = WoT + (size_t)(n0 + 16 * j + ln) * 1024 + k + hi * 8;
      bf[j].q[0] = *(const uint4*)p;
      bf[j].q[1] = *(const uint4*)(p + 16);
    }
    for (int i = 0; i < 4; ++i)
      for (int j = 0; j < 4; ++j)
        acc[i][j] = wmma_bf16(af[i], bf[j], acc[i][j]);
  }

  for (int j = 0; j < 4; ++j) {
    int n = n0 + 16 * j + ln;
    float bv_ = bo[n];
    for (int i = 0; i < 4; ++i) {
      for (int r = 0; r < 8; ++r) {
        int row = m0 + 16 * i + r + hi * 8;
        y[(size_t)row * 1024 + n] = acc[i][j][r] + bv_;
      }
    }
  }
}

// ---------------------------------------------------------------------------
// Launch
// ---------------------------------------------------------------------------
extern "C" void kernel_launch(void* const* d_in, const int* in_sizes, int n_in,
                              void* d_out, int out_size, void* d_ws, size_t ws_size,
                              hipStream_t stream) {
  const float* x   = (const float*)d_in[0];
  const int*   mod = (const int*)d_in[1];
  const float* Wq  = (const float*)d_in[2];
  const float* bq  = (const float*)d_in[3];
  const float* Wk  = (const float*)d_in[4];
  const float* bk  = (const float*)d_in[5];
  const float* Wv  = (const float*)d_in[6];
  const float* bv  = (const float*)d_in[7];
  const float* Wo  = (const float*)d_in[8];
  const float* bo  = (const float*)d_in[9];

  char* ws = (char*)d_ws;
  unsigned short* xb   = (unsigned short*)(ws + OFF_XB);
  unsigned short* WT   = (unsigned short*)(ws + OFF_WQKV);
  float*          bias = (float*)(ws + OFF_BIAS);
  unsigned short* WoT  = (unsigned short*)(ws + OFF_WOT);
  unsigned short* Qb   = (unsigned short*)(ws + OFF_Q);
  unsigned short* Kb   = (unsigned short*)(ws + OFF_K);
  unsigned short* Vt   = (unsigned short*)(ws + OFF_VT);
  unsigned short* ob   = (unsigned short*)(ws + OFF_OB);
  float*          y    = (float*)d_out;

  cvt_x_kernel<<<(8192 * 1024 + 255) / 256, 256, 0, stream>>>(x, xb, 8192 * 1024);
  build_wqkvT_kernel<<<(3072 * 1024 + 255) / 256, 256, 0, stream>>>(Wq, Wk, Wv, WT);
  build_bias_kernel<<<(3072 + 255) / 256, 256, 0, stream>>>(bq, bk, bv, bias);
  build_woT_kernel<<<(1024 * 1024 + 255) / 256, 256, 0, stream>>>(Wo, WoT);

  qkv_gemm_kernel<<<dim3(64, 12), 256, 0, stream>>>(xb, WT, bias, Qb, Kb, Vt);
  attn_kernel<<<128, 256, 0, stream>>>(Qb, Kb, Vt, mod, ob);
  out_gemm_kernel<<<dim3(64, 4), 256, 0, stream>>>(ob, WoT, bo, y);
}
